// Int4MLP_40913858462384
// MI455X (gfx1250) — matrix operand order repaired and checked by reading, not compiled, not run on hardware
//
#include <hip/hip_runtime.h>

#define H_DIM 4096
#define I_DIM 14336
#define M_DIM 8192
#define GS    128

typedef __attribute__((ext_vector_type(16))) __bf16 v16bf;
typedef __attribute__((ext_vector_type(8)))  float  v8f;

union FragAB { v16bf v; uint4 q[2]; };
union FragC  { v8f   v; float f[8]; };
union BF8    { __bf16 b[8]; uint4 q; };

// ---------------------------------------------------------------------------
// Async global->LDS copy (gfx1250 GLOBAL_LOAD_ASYNC_TO_LDS_B128, ASYNCcnt).
// ---------------------------------------------------------------------------
#if defined(__HIP_DEVICE_COMPILE__) && __has_builtin(__builtin_amdgcn_global_load_async_to_lds_b128)
#define ASYNC_LDS 1
typedef int v4i_vs __attribute__((vector_size(16)));
typedef __attribute__((address_space(1))) v4i_vs* gq_ptr;
typedef __attribute__((address_space(3))) v4i_vs* lq_ptr;
__device__ __forceinline__ void async_copy16(void* lds_dst, const void* gsrc) {
    __builtin_amdgcn_global_load_async_to_lds_b128(
        (gq_ptr)gsrc, (lq_ptr)lds_dst, 0, 0);
}
__device__ __forceinline__ void async_wait0() {
#if __has_builtin(__builtin_amdgcn_s_wait_asynccnt)
    __builtin_amdgcn_s_wait_asynccnt(0);
#else
    asm volatile("s_wait_asynccnt 0x0" ::: "memory");
#endif
}
#else
#define ASYNC_LDS 0
__device__ __forceinline__ void async_copy16(void* lds_dst, const void* gsrc) {
    *(uint4*)lds_dst = *(const uint4*)gsrc;
}
__device__ __forceinline__ void async_wait0() {}
#endif

// ---------------------------------------------------------------------------
// Pass 0: x fp32 -> bf16 (once)
// ---------------------------------------------------------------------------
__global__ __launch_bounds__(256) void cvt_x(const float* __restrict__ x,
                                             __bf16* __restrict__ xb) {
    size_t i = ((size_t)blockIdx.x * 256 + threadIdx.x) * 8;
    float4 a0 = *(const float4*)(x + i);
    float4 a1 = *(const float4*)(x + i + 4);
    BF8 o;
    o.b[0] = (__bf16)a0.x; o.b[1] = (__bf16)a0.y; o.b[2] = (__bf16)a0.z; o.b[3] = (__bf16)a0.w;
    o.b[4] = (__bf16)a1.x; o.b[5] = (__bf16)a1.y; o.b[6] = (__bf16)a1.z; o.b[7] = (__bf16)a1.w;
    *(uint4*)(xb + i) = o.q;
}

// ---------------------------------------------------------------------------
// Kernel 1: hidden[M,I] (bf16) = silu(X @ deq(gate)) * (X @ deq(up))
// Block tile 128x128x32, 8 waves (2 in M x 4 in N), wave tile 64x32.
// Double-buffered LDS; A staged async; weights dequantized on the fly.
// ---------------------------------------------------------------------------
__global__ __launch_bounds__(256) void gemm_gate_up(
    const __bf16* __restrict__ xb,
    const int* __restrict__ gw, const int* __restrict__ gz, const float* __restrict__ gs,
    const int* __restrict__ uw, const int* __restrict__ uz, const float* __restrict__ us,
    __bf16* __restrict__ hidden)
{
    __shared__ __bf16 Xs[2][128 * 32];   // [m][k]
    __shared__ __bf16 Gsh[2][128 * 32];  // [n][k]
    __shared__ __bf16 Ush[2][128 * 32];  // [n][k]

    const int n0   = blockIdx.x * 128;
    const int m0   = blockIdx.y * 128;
    const int tid  = threadIdx.x;
    const int lane = tid & 31;
    const int wid  = tid >> 5;
    const int wave_m = wid >> 2;   // 0..1 -> 64 rows each
    const int wave_n = wid & 3;    // 0..3 -> 32 cols each

    FragC accG[4][2], accU[4][2];
    #pragma unroll
    for (int i = 0; i < 4; i++)
        #pragma unroll
        for (int j = 0; j < 2; j++)
            #pragma unroll
            for (int r = 0; r < 8; r++) { accG[i][j].f[r] = 0.f; accU[i][j].f[r] = 0.f; }

    const int xr = tid >> 1;          // A row 0..127
    const int xc = (tid & 1) * 16;    // A col {0,16}
    const int wc = (tid & 31) * 4;    // W col 0..124
    const int wk = (tid >> 5) * 4;    // W k-row 0..28
    const int half = lane >> 4;
    const int lm   = lane & 15;

    auto stage = [&](int buf, int k0) {
        // A tile: 128x32 bf16, 2x16B per thread, async global->LDS
        const __bf16* asrc = xb + (size_t)(m0 + xr) * H_DIM + k0 + xc;
        async_copy16(&Xs[buf][xr * 32 + xc],     asrc);
        async_copy16(&Xs[buf][xr * 32 + xc + 8], asrc + 8);
        const int g = k0 / GS;   // 32-slab lies in one quant group
        {
            float s4[4], zs4[4];
            #pragma unroll
            for (int c = 0; c < 4; c++) {
                float s = gs[(size_t)g * I_DIM + n0 + wc + c];
                int   z = gz[(size_t)g * I_DIM + n0 + wc + c];
                s4[c] = s; zs4[c] = s * (float)z;
            }
            #pragma unroll
            for (int r = 0; r < 4; r++) {
                int4 w = *(const int4*)(gw + (size_t)(k0 + wk + r) * I_DIM + n0 + wc);
                Gsh[buf][(wc + 0) * 32 + wk + r] = (__bf16)((float)w.x * s4[0] - zs4[0]);
                Gsh[buf][(wc + 1) * 32 + wk + r] = (__bf16)((float)w.y * s4[1] - zs4[1]);
                Gsh[buf][(wc + 2) * 32 + wk + r] = (__bf16)((float)w.z * s4[2] - zs4[2]);
                Gsh[buf][(wc + 3) * 32 + wk + r] = (__bf16)((float)w.w * s4[3] - zs4[3]);
            }
        }
        {
            float s4[4], zs4[4];
            #pragma unroll
            for (int c = 0; c < 4; c++) {
                float s = us[(size_t)g * I_DIM + n0 + wc + c];
                int   z = uz[(size_t)g * I_DIM + n0 + wc + c];
                s4[c] = s; zs4[c] = s * (float)z;
            }
            #pragma unroll
            for (int r = 0; r < 4; r++) {
                int4 w = *(const int4*)(uw + (size_t)(k0 + wk + r) * I_DIM + n0 + wc);
                Ush[buf][(wc + 0) * 32 + wk + r] = (__bf16)((float)w.x * s4[0] - zs4[0]);
                Ush[buf][(wc + 1) * 32 + wk + r] = (__bf16)((float)w.y * s4[1] - zs4[1]);
                Ush[buf][(wc + 2) * 32 + wk + r] = (__bf16)((float)w.z * s4[2] - zs4[2]);
                Ush[buf][(wc + 3) * 32 + wk + r] = (__bf16)((float)w.w * s4[3] - zs4[3]);
            }
        }
    };

    stage(0, 0);
    async_wait0();
    __syncthreads();

    for (int k0 = 0; k0 < H_DIM; k0 += 32) {
        const int p = (k0 >> 5) & 1;
        if (k0 + 32 < H_DIM) stage(1 - p, k0 + 32);

        // A frags (64 rows): lanes 0-15 K{0..7,16..23}; lanes 16-31 K{8..15,24..31}
        FragAB afr[4];
        #pragma unroll
        for (int mi = 0; mi < 4; mi++) {
            const __bf16* src = &Xs[p][(wave_m * 64 + mi * 16 + lm) * 32 + (half ? 8 : 0)];
            afr[mi].q[0] = *(const uint4*)(src);
            afr[mi].q[1] = *(const uint4*)(src + 16);
        }
        // B frags (32 cols): lanes 0-15 col lm K{0..7,16..23}; lanes 16-31 K{8..15,24..31}
        FragAB bg[2], bu[2];
        #pragma unroll
        for (int ni = 0; ni < 2; ni++) {
            const int nl = wave_n * 32 + ni * 16 + lm;
            const __bf16* sg = &Gsh[p][nl * 32 + (half ? 8 : 0)];
            bg[ni].q[0] = *(const uint4*)(sg);
            bg[ni].q[1] = *(const uint4*)(sg + 16);
            const __bf16* su = &Ush[p][nl * 32 + (half ? 8 : 0)];
            bu[ni].q[0] = *(const uint4*)(su);
            bu[ni].q[1] = *(const uint4*)(su + 16);
        }

        #pragma unroll
        for (int mi = 0; mi < 4; mi++)
            #pragma unroll
            for (int ni = 0; ni < 2; ni++) {
                accG[mi][ni].v = __builtin_amdgcn_wmma_f32_16x16x32_bf16(
                    false, afr[mi].v, false, bg[ni].v, (short)0, accG[mi][ni].v, false, false);
                accU[mi][ni].v = __builtin_amdgcn_wmma_f32_16x16x32_bf16(
                    false, afr[mi].v, false, bu[ni].v, (short)0, accU[mi][ni].v, false, false);
            }

        async_wait0();
        __syncthreads();
    }

    // Epilogue: silu(g)*u with v_rcp
    #pragma unroll
    for (int mi = 0; mi < 4; mi++)
        #pragma unroll
        for (int ni = 0; ni < 2; ni++)
            #pragma unroll
            for (int r = 0; r < 8; r++) {
                float gv = accG[mi][ni].f[r];
                float uv = accU[mi][ni].f[r];
                float sig = __builtin_amdgcn_rcpf(1.f + __expf(-gv));
                float hv = gv * sig * uv;
                int m = m0 + wave_m * 64 + mi * 16 + r + half * 8;
                int n = n0 + wave_n * 32 + ni * 16 + lm;
                hidden[(size_t)m * I_DIM + n] = (__bf16)hv;
            }
}

// ---------------------------------------------------------------------------
// Kernel 2: out[M,H] (fp32) = hidden[M,I] (bf16) @ deq(down)[I,H]
// Same tiling: block 128x128x32, wave tile 64x32.
// ---------------------------------------------------------------------------
__global__ __launch_bounds__(256) void gemm_down(
    const __bf16* __restrict__ hidden,
    const int* __restrict__ dw, const int* __restrict__ dz, const float* __restrict__ dsc,
    float* __restrict__ out)
{
    __shared__ __bf16 As[2][128 * 32];
    __shared__ __bf16 Wsh[2][128 * 32];

    const int n0   = blockIdx.x * 128;
    const int m0   = blockIdx.y * 128;
    const int tid  = threadIdx.x;
    const int lane = tid & 31;
    const int wid  = tid >> 5;
    const int wave_m = wid >> 2;
    const int wave_n = wid & 3;

    FragC acc[4][2];
    #pragma unroll
    for (int i = 0; i < 4; i++)
        #pragma unroll
        for (int j = 0; j < 2; j++)
            #pragma unroll
            for (int r = 0; r < 8; r++) acc[i][j].f[r] = 0.f;

    const int xr = tid >> 1;
    const int xc = (tid & 1) * 16;
    const int wc = (tid & 31) * 4;
    const int wk = (tid >> 5) * 4;
    const int half = lane >> 4;
    const int lm   = lane & 15;

    auto stage = [&](int buf, int k0) {
        const __bf16* asrc = hidden + (size_t)(m0 + xr) * I_DIM + k0 + xc;
        async_copy16(&As[buf][xr * 32 + xc],     asrc);
        async_copy16(&As[buf][xr * 32 + xc + 8], asrc + 8);
        const int g = k0 / GS;
        float s4[4], zs4[4];
        #pragma unroll
        for (int c = 0; c < 4; c++) {
            float s = dsc[(size_t)g * H_DIM + n0 + wc + c];
            int   z = dz[(size_t)g * H_DIM + n0 + wc + c];
            s4[c] = s; zs4[c] = s * (float)z;
        }
        #pragma unroll
        for (int r = 0; r < 4; r++) {
            int4 w = *(const int4*)(dw + (size_t)(k0 + wk + r) * H_DIM + n0 + wc);
            Wsh[buf][(wc + 0) * 32 + wk + r] = (__bf16)((float)w.x * s4[0] - zs4[0]);
            Wsh[buf][(wc + 1) * 32 + wk + r] = (__bf16)((float)w.y * s4[1] - zs4[1]);
            Wsh[buf][(wc + 2) * 32 + wk + r] = (__bf16)((float)w.z * s4[2] - zs4[2]);
            Wsh[buf][(wc + 3) * 32 + wk + r] = (__bf16)((float)w.w * s4[3] - zs4[3]);
        }
    };

    stage(0, 0);
    async_wait0();
    __syncthreads();

    for (int k0 = 0; k0 < I_DIM; k0 += 32) {
        const int p = (k0 >> 5) & 1;
        if (k0 + 32 < I_DIM) stage(1 - p, k0 + 32);

        FragAB afr[4], bfr[2];
        #pragma unroll
        for (int mi = 0; mi < 4; mi++) {
            const __bf16* src = &As[p][(wave_m * 64 + mi * 16 + lm) * 32 + (half ? 8 : 0)];
            afr[mi].q[0] = *(const uint4*)(src);
            afr[mi].q[1] = *(const uint4*)(src + 16);
        }
        #pragma unroll
        for (int ni = 0; ni < 2; ni++) {
            const __bf16* src = &Wsh[p][(wave_n * 32 + ni * 16 + lm) * 32 + (half ? 8 : 0)];
            bfr[ni].q[0] = *(const uint4*)(src);
            bfr[ni].q[1] = *(const uint4*)(src + 16);
        }

        #pragma unroll
        for (int mi = 0; mi < 4; mi++)
            #pragma unroll
            for (int ni = 0; ni < 2; ni++)
                acc[mi][ni].v = __builtin_amdgcn_wmma_f32_16x16x32_bf16(
                    false, afr[mi].v, false, bfr[ni].v, (short)0, acc[mi][ni].v, false, false);

        async_wait0();
        __syncthreads();
    }

    #pragma unroll
    for (int mi = 0; mi < 4; mi++)
        #pragma unroll
        for (int ni = 0; ni < 2; ni++)
            #pragma unroll
            for (int r = 0; r < 8; r++) {
                int m = m0 + wave_m * 64 + mi * 16 + r + half * 8;
                int n = n0 + wave_n * 32 + ni * 16 + lm;
                out[(size_t)m * H_DIM + n] = acc[mi][ni].f[r];
            }
}

// ---------------------------------------------------------------------------
extern "C" void kernel_launch(void* const* d_in, const int* in_sizes, int n_in,
                              void* d_out, int out_size, void* d_ws, size_t ws_size,
                              hipStream_t stream) {
    const float* x   = (const float*)d_in[0];
    const int*   gw  = (const int*)  d_in[1];
    const int*   gz  = (const int*)  d_in[2];
    const float* gs  = (const float*)d_in[3];
    const int*   uw  = (const int*)  d_in[4];
    const int*   uz  = (const int*)  d_in[5];
    const float* us  = (const float*)d_in[6];
    const int*   dw  = (const int*)  d_in[7];
    const int*   dz  = (const int*)  d_in[8];
    const float* dsc = (const float*)d_in[9];

    const size_t hidden_bytes = (size_t)M_DIM * I_DIM * 2;  // 235 MB
    __bf16* hidden = (__bf16*)d_ws;
    __bf16* xbf    = (__bf16*)((char*)d_ws + hidden_bytes); // +67 MB
    float*  out    = (float*)d_out;

    cvt_x<<<(M_DIM * (size_t)H_DIM) / (256 * 8), 256, 0, stream>>>(x, xbf);

    dim3 g1(I_DIM / 128, M_DIM / 128);  // 112 x 64
    gemm_gate_up<<<g1, 256, 0, stream>>>(xbf, gw, gz, gs, uw, uz, us, hidden);

    dim3 g2(H_DIM / 128, M_DIM / 128);  // 32 x 64
    gemm_down<<<g2, 256, 0, stream>>>(hidden, dw, dz, dsc, out);
}
